// ArcFaceSingleDomain_50139448213649
// MI455X (gfx1250) — matrix operand order repaired and checked by reading, not compiled, not run on hardware
//
#include <hip/hip_runtime.h>

typedef __attribute__((ext_vector_type(16))) __bf16 v16bf;
typedef __attribute__((ext_vector_type(8)))  float  v8f;

#define EMBED   512
#define NLAB    50000
#define BATCH   1024
#define NTOT    150000                  // NLAB * K(=3) weight rows
#define OUT_STRIDE ((long)BATCH * NLAB) // 51,200,000

#define COS_M  0.95533648912560601964f
#define SIN_M  0.29552020666133957510f
#define TH_    (-0.95533648912560601964f)
#define MM_    0.08865606199840187254f
#define SCALE_ 30.0f

// Tiled bf16 weight workspace geometry: [col_tile][tn(3)][kb(16)][lh(2)][l15(16)][e(16)]
#define WT_COLTILE_STRIDE 24576         // 48 rows * 512
#define WT_TN_STRIDE      8192          // 16*2*16*16
#define WT_KB_STRIDE      512
#define WT_LH_STRIDE      256

// Native bf16 conversion: let the backend pick the hardware cvt if present.
__device__ __forceinline__ unsigned pk2(float a, float b) {
  union { unsigned u; __bf16 h[2]; } r;
  r.h[0] = (__bf16)a;
  r.h[1] = (__bf16)b;
  return r.u;
}
__device__ __forceinline__ float wred(float s) {
#pragma unroll
  for (int off = 16; off > 0; off >>= 1) s += __shfl_xor(s, off, 32);
  return s;
}

// ---- normalize x rows -> bf16, row-major (one wave per row) --------------
__global__ __launch_bounds__(256) void k_norm_x(const float* __restrict__ x,
                                                unsigned short* __restrict__ xnb) {
  int wave = threadIdx.x >> 5, lane = threadIdx.x & 31;
  int row  = blockIdx.x * 8 + wave;
  const float4* q = (const float4*)(x + (size_t)row * EMBED + lane * 16);
  float4 f0 = q[0], f1 = q[1], f2 = q[2], f3 = q[3];
  float s = f0.x*f0.x + f0.y*f0.y + f0.z*f0.z + f0.w*f0.w
          + f1.x*f1.x + f1.y*f1.y + f1.z*f1.z + f1.w*f1.w
          + f2.x*f2.x + f2.y*f2.y + f2.z*f2.z + f2.w*f2.w
          + f3.x*f3.x + f3.y*f3.y + f3.z*f3.z + f3.w*f3.w;
  s = wred(s);
  float inv = rsqrtf(s + 1e-12f);
  uint4 o0, o1;
  o0.x = pk2(f0.x*inv, f0.y*inv); o0.y = pk2(f0.z*inv, f0.w*inv);
  o0.z = pk2(f1.x*inv, f1.y*inv); o0.w = pk2(f1.z*inv, f1.w*inv);
  o1.x = pk2(f2.x*inv, f2.y*inv); o1.y = pk2(f2.z*inv, f2.w*inv);
  o1.z = pk2(f3.x*inv, f3.y*inv); o1.w = pk2(f3.z*inv, f3.w*inv);
  uint4* dst = (uint4*)(xnb + (size_t)row * EMBED + lane * 16);
  dst[0] = o0; dst[1] = o1;
}

// ---- normalize W rows -> bf16 in WMMA-B swizzled layout (one wave/row) ---
__global__ __launch_bounds__(256) void k_norm_w_pre(const float* __restrict__ w,
                                                    unsigned short* __restrict__ wnb) {
  int wave = threadIdx.x >> 5, lane = threadIdx.x & 31;
  int row  = blockIdx.x * 8 + wave;
  const float4* q = (const float4*)(w + (size_t)row * EMBED + lane * 16);
  float4 f0 = q[0], f1 = q[1], f2 = q[2], f3 = q[3];
  float s = f0.x*f0.x + f0.y*f0.y + f0.z*f0.z + f0.w*f0.w
          + f1.x*f1.x + f1.y*f1.y + f1.z*f1.z + f1.w*f1.w
          + f2.x*f2.x + f2.y*f2.y + f2.z*f2.z + f2.w*f2.w
          + f3.x*f3.x + f3.y*f3.y + f3.z*f3.z + f3.w*f3.w;
  s = wred(s);
  float inv = rsqrtf(s + 1e-12f);
  // lane covers k = lane*16 .. lane*16+15  =>  kb = lane>>1, lh = lane&1
  int col_tile = row / 48, r48 = row % 48;
  int tn = r48 >> 4, l15 = r48 & 15;
  int kb = lane >> 1, lh = lane & 1;
  size_t dst = (size_t)col_tile * WT_COLTILE_STRIDE + tn * WT_TN_STRIDE
             + kb * WT_KB_STRIDE + lh * WT_LH_STRIDE + l15 * 16;
  uint4 o0, o1;
  o0.x = pk2(f0.x*inv, f0.y*inv); o0.y = pk2(f0.z*inv, f0.w*inv);
  o0.z = pk2(f1.x*inv, f1.y*inv); o0.w = pk2(f1.z*inv, f1.w*inv);
  o1.x = pk2(f2.x*inv, f2.y*inv); o1.y = pk2(f2.z*inv, f2.w*inv);
  o1.z = pk2(f3.x*inv, f3.y*inv); o1.w = pk2(f3.z*inv, f3.w*inv);
  uint4* d = (uint4*)(wnb + dst);
  d[0] = o0; d[1] = o1;
}

// ---- weight row inverse norms only (fallback when ws is small) -----------
__global__ __launch_bounds__(256) void k_norm_w_inv(const float* __restrict__ w,
                                                    float* __restrict__ invw) {
  int wave = threadIdx.x >> 5, lane = threadIdx.x & 31;
  int row  = blockIdx.x * 8 + wave;
  const float4* q = (const float4*)(w + (size_t)row * EMBED + lane * 16);
  float4 f0 = q[0], f1 = q[1], f2 = q[2], f3 = q[3];
  float s = f0.x*f0.x + f0.y*f0.y + f0.z*f0.z + f0.w*f0.w
          + f1.x*f1.x + f1.y*f1.y + f1.z*f1.z + f1.w*f1.w
          + f2.x*f2.x + f2.y*f2.y + f2.z*f2.z + f2.w*f2.w
          + f3.x*f3.x + f3.y*f3.y + f3.z*f3.z + f3.w*f3.w;
  s = wred(s);
  if (lane == 0) invw[row] = rsqrtf(s + 1e-12f);
}

// ---- shared epilogue: max over K=3 + ArcFace margin ----------------------
__device__ __forceinline__ void epilogue(float* __restrict__ L,
                                         const int* __restrict__ labels,
                                         float* __restrict__ out,
                                         int m0, int col_tile, int lane) {
  int label0 = col_tile * 16;
  float* outL = out;
  float* outM = out + OUT_STRIDE;
#pragma unroll
  for (int i = 0; i < 8; ++i) {
    int idx = i * 32 + lane;
    int r = idx >> 4, c = idx & 15;
    const float* p = L + r * 48 + c * 3;
    float cosv = fmaxf(p[0], fmaxf(p[1], p[2]));
    float sine = sqrtf(fminf(fmaxf(1.0f - cosv * cosv, 0.0f), 1.0f));
    float phi  = cosv * COS_M - sine * SIN_M;
    phi = (cosv > TH_) ? phi : (cosv - MM_);
    int gl  = label0 + c;
    int lab = labels[m0 + r];
    float om = ((lab == gl) ? phi : cosv) * SCALE_;
    long o = (long)(m0 + r) * NLAB + gl;
    outL[o] = cosv;
    outM[o] = om;
  }
}

// ---- main GEMM: pre-normalized swizzled bf16 weights ---------------------
// grid = 3125 col-tiles * 8 row-blocks; block = 256 (8 waves)
__global__ __launch_bounds__(256) void k_gemm_pre(const unsigned short* __restrict__ xnb,
                                                  const unsigned short* __restrict__ wnb,
                                                  const int* __restrict__ labels,
                                                  float* __restrict__ out) {
  __shared__ float lds[8][16 * 48];
  int wave = threadIdx.x >> 5, lane = threadIdx.x & 31;
  int col_tile = blockIdx.x >> 3;
  int rb       = blockIdx.x & 7;
  int m0 = (rb * 8 + wave) * 16;
  int lh = lane >> 4, l15 = lane & 15;

  const unsigned short* xrow = xnb + (size_t)(m0 + l15) * EMBED;
  const unsigned short* wt   = wnb + (size_t)col_tile * WT_COLTILE_STRIDE
                                   + lh * WT_LH_STRIDE + l15 * 16;

  v8f acc0 = {}, acc1 = {}, acc2 = {};

#pragma unroll 4
  for (int kb = 0; kb < EMBED / 32; ++kb) {
    union { v16bf v; uint4 q[2]; } af, b0, b1, b2;
    int ka = kb * 32 + lh * 16;
    af.q[0] = *(const uint4*)(xrow + ka);
    af.q[1] = *(const uint4*)(xrow + ka + 8);
    const uint4* p0 = (const uint4*)(wt + kb * WT_KB_STRIDE);
    const uint4* p1 = (const uint4*)(wt + kb * WT_KB_STRIDE + WT_TN_STRIDE);
    const uint4* p2 = (const uint4*)(wt + kb * WT_KB_STRIDE + 2 * WT_TN_STRIDE);
    b0.q[0] = p0[0]; b0.q[1] = p0[1];
    b1.q[0] = p1[0]; b1.q[1] = p1[1];
    b2.q[0] = p2[0]; b2.q[1] = p2[1];
    acc0 = __builtin_amdgcn_wmma_f32_16x16x32_bf16(false, af.v, false, b0.v, (short)0, acc0, false, false);
    acc1 = __builtin_amdgcn_wmma_f32_16x16x32_bf16(false, af.v, false, b1.v, (short)0, acc1, false, false);
    acc2 = __builtin_amdgcn_wmma_f32_16x16x32_bf16(false, af.v, false, b2.v, (short)0, acc2, false, false);
  }

  float* L = lds[wave];
  int rbase = lh * 8;
#pragma unroll
  for (int v = 0; v < 8; ++v) {
    int r = rbase + v;
    L[r * 48 + l15]      = acc0[v];
    L[r * 48 + 16 + l15] = acc1[v];
    L[r * 48 + 32 + l15] = acc2[v];
  }
  __syncthreads();
  epilogue(L, labels, out, m0, col_tile, lane);
}

// ---- fallback GEMM: on-the-fly f32->bf16 + invw scaling ------------------
__device__ __forceinline__ v16bf ldb(const float* __restrict__ p) {
  union { v16bf v; unsigned u[8]; } r;
  const float4* q = (const float4*)p;
  float4 f0 = q[0], f1 = q[1], f2 = q[2], f3 = q[3];
  r.u[0] = pk2(f0.x, f0.y); r.u[1] = pk2(f0.z, f0.w);
  r.u[2] = pk2(f1.x, f1.y); r.u[3] = pk2(f1.z, f1.w);
  r.u[4] = pk2(f2.x, f2.y); r.u[5] = pk2(f2.z, f2.w);
  r.u[6] = pk2(f3.x, f3.y); r.u[7] = pk2(f3.z, f3.w);
  return r.v;
}

__global__ __launch_bounds__(256) void k_gemm_fly(const unsigned short* __restrict__ xnb,
                                                  const float* __restrict__ w,
                                                  const float* __restrict__ invw,
                                                  const int* __restrict__ labels,
                                                  float* __restrict__ out) {
  __shared__ float lds[8][16 * 48];
  int wave = threadIdx.x >> 5, lane = threadIdx.x & 31;
  int col_tile = blockIdx.x >> 3;
  int rb       = blockIdx.x & 7;
  int m0 = (rb * 8 + wave) * 16;
  int n0 = col_tile * 48;
  int lh = lane >> 4, l15 = lane & 15;

  const unsigned short* xrow = xnb + (size_t)(m0 + l15) * EMBED;
  const float* w0 = w + (size_t)(n0 + l15) * EMBED;
  const float* w1 = w0 + 16 * EMBED;
  const float* w2 = w0 + 32 * EMBED;

  v8f acc0 = {}, acc1 = {}, acc2 = {};
#pragma unroll 2
  for (int kb = 0; kb < EMBED / 32; ++kb) {
    union { v16bf v; uint4 q[2]; } af;
    int ka = kb * 32 + lh * 16;
    af.q[0] = *(const uint4*)(xrow + ka);
    af.q[1] = *(const uint4*)(xrow + ka + 8);
    int kw = kb * 32 + lh * 16;
    v16bf b0 = ldb(w0 + kw);
    v16bf b1 = ldb(w1 + kw);
    v16bf b2 = ldb(w2 + kw);
    acc0 = __builtin_amdgcn_wmma_f32_16x16x32_bf16(false, af.v, false, b0, (short)0, acc0, false, false);
    acc1 = __builtin_amdgcn_wmma_f32_16x16x32_bf16(false, af.v, false, b1, (short)0, acc1, false, false);
    acc2 = __builtin_amdgcn_wmma_f32_16x16x32_bf16(false, af.v, false, b2, (short)0, acc2, false, false);
  }
  float s0 = invw[n0 + l15];
  float s1 = invw[n0 + 16 + l15];
  float s2 = invw[n0 + 32 + l15];
  float* L = lds[wave];
  int rbase = lh * 8;
#pragma unroll
  for (int v = 0; v < 8; ++v) {
    int r = rbase + v;
    L[r * 48 + l15]      = acc0[v] * s0;
    L[r * 48 + 16 + l15] = acc1[v] * s1;
    L[r * 48 + 32 + l15] = acc2[v] * s2;
  }
  __syncthreads();
  epilogue(L, labels, out, m0, col_tile, lane);
}

extern "C" void kernel_launch(void* const* d_in, const int* in_sizes, int n_in,
                              void* d_out, int out_size, void* d_ws, size_t ws_size,
                              hipStream_t stream) {
  const float* x      = (const float*)d_in[0];
  const int*   labels = (const int*)d_in[1];
  const float* w      = (const float*)d_in[2];
  float* out = (float*)d_out;

  const size_t xnb_bytes = (size_t)BATCH * EMBED * 2;   // 1 MB
  const size_t wnb_bytes = (size_t)NTOT * EMBED * 2;    // 153.6 MB
  unsigned short* xnb = (unsigned short*)d_ws;

  hipLaunchKernelGGL(k_norm_x, dim3(BATCH / 8), dim3(256), 0, stream, x, xnb);

  if (ws_size >= xnb_bytes + wnb_bytes) {
    unsigned short* wnb = (unsigned short*)((char*)d_ws + xnb_bytes);
    hipLaunchKernelGGL(k_norm_w_pre, dim3(NTOT / 8), dim3(256), 0, stream, w, wnb);
    hipLaunchKernelGGL(k_gemm_pre, dim3((NLAB / 16) * 8), dim3(256), 0, stream,
                       xnb, wnb, labels, out);
  } else {
    float* invw = (float*)((char*)d_ws + xnb_bytes);
    hipLaunchKernelGGL(k_norm_w_inv, dim3(NTOT / 8), dim3(256), 0, stream, w, invw);
    hipLaunchKernelGGL(k_gemm_fly, dim3((NLAB / 16) * 8), dim3(256), 0, stream,
                       xnb, w, invw, labels, out);
  }
}
